// TransformerBlock_78185584656511
// MI455X (gfx1250) — matrix operand order repaired and checked by reading, not compiled, not run on hardware
//
#include <hip/hip_runtime.h>
#include <hip/hip_bf16.h>

// ---------------------------------------------------------------------------
// Types for gfx1250 WMMA
// ---------------------------------------------------------------------------
typedef __bf16 v16bf __attribute__((ext_vector_type(16)));
typedef __bf16 v8bf  __attribute__((ext_vector_type(8)));
typedef float  v8f   __attribute__((ext_vector_type(8)));

union V8F { v8f v; float f[8]; };

__device__ inline v16bf cat16(v8bf lo, v8bf hi) {
  return __builtin_shufflevector(lo, hi, 0,1,2,3,4,5,6,7,8,9,10,11,12,13,14,15);
}

__device__ inline unsigned short f2bf(float f) {
  union { float f; unsigned int u; } c; c.f = f;
  unsigned int u = c.u;
  unsigned int r = (u + 0x7FFFu + ((u >> 16) & 1u)) >> 16;   // round-to-nearest-even
  return (unsigned short)r;
}
__device__ inline float bf2f(unsigned short u) {
  union { unsigned int i; float f; } c; c.i = ((unsigned int)u) << 16; return c.f;
}

// Low 32 bits of a generic shared-memory pointer == LDS byte offset
// (CDNA5 ISA 10.2: an LDS-aperture address uses its low 32 bits as the LDS address).
__device__ inline unsigned lds_off(const void* p) {
  return (unsigned)(unsigned long long)p;
}

// gfx1250 async copy: global -> LDS, 16B per lane, tracked by ASYNCcnt.
__device__ inline void async_load_b128(unsigned lds_dst, const void* gsrc) {
  asm volatile("global_load_async_to_lds_b128 %0, %1, off"
               :: "v"(lds_dst), "v"(gsrc) : "memory");
}
__device__ inline void wait_asynccnt0() {
  asm volatile("s_wait_asynccnt 0x0" ::: "memory");
}

// ---------------------------------------------------------------------------
// Elementwise: f32 -> bf16 (weights, V tensor)
// ---------------------------------------------------------------------------
__global__ void f32_to_bf16_kernel(const float* __restrict__ in,
                                   unsigned short* __restrict__ out, size_t n) {
  size_t i = (size_t)blockIdx.x * blockDim.x + threadIdx.x;
  size_t stride = (size_t)gridDim.x * blockDim.x;
  for (; i < n; i += stride) out[i] = f2bf(in[i]);
}

// ---------------------------------------------------------------------------
// RMSNorm (one block of 256 threads per row) -> bf16
// ---------------------------------------------------------------------------
__global__ __launch_bounds__(256) void rmsnorm_bf16_kernel(
    const float* __restrict__ in, unsigned short* __restrict__ out, int D_) {
  __shared__ float red[256];
  int row = blockIdx.x, t = threadIdx.x;
  const float* x = in + (size_t)row * D_;
  float ss = 0.f;
  for (int i = t; i < D_; i += 256) { float v = x[i]; ss += v * v; }
  red[t] = ss; __syncthreads();
  for (int s = 128; s > 0; s >>= 1) { if (t < s) red[t] += red[t + s]; __syncthreads(); }
  float inv = rsqrtf(red[0] / (float)D_ + 1e-8f);
  unsigned short* o = out + (size_t)row * D_;
  for (int i = t; i < D_; i += 256) o[i] = f2bf(x[i] * inv);
}

// ---------------------------------------------------------------------------
// RoPE: one wave per (token, head) 64-dim vector; f32 in -> bf16 out
// ---------------------------------------------------------------------------
__global__ __launch_bounds__(256) void rope_bf16_kernel(
    const float* __restrict__ in, unsigned short* __restrict__ out,
    int nHeads, int S_) {
  int vec  = blockIdx.x * (blockDim.x >> 5) + (threadIdx.x >> 5);
  int lane = threadIdx.x & 31;
  int token = vec / nHeads;
  int pos = token % S_;
  const float* src = in + (size_t)vec * 64;
  unsigned short* dst = out + (size_t)vec * 64;
  float invf = __powf(10000.0f, -(float)lane / 32.0f);
  float ang = (float)pos * invf;
  float s, c;
  __sincosf(ang, &s, &c);
  float x0 = src[2 * lane], x1 = src[2 * lane + 1];
  dst[2 * lane]     = f2bf(x0 * c - x1 * s);
  dst[2 * lane + 1] = f2bf(x0 * s + x1 * c);
}

// ---------------------------------------------------------------------------
// SwiGLU: ff[r, 0:HID] * silu(ff[r, HID:2HID]) -> bf16
// ---------------------------------------------------------------------------
__global__ void swiglu_kernel(const unsigned short* __restrict__ ff,
                              unsigned short* __restrict__ out,
                              int HID_, size_t total) {
  size_t i = (size_t)blockIdx.x * blockDim.x + threadIdx.x;
  size_t stride = (size_t)gridDim.x * blockDim.x;
  for (; i < total; i += stride) {
    size_t r = i / HID_; int n = (int)(i % HID_);
    float x1 = bf2f(ff[r * (size_t)(2 * HID_) + n]);
    float x2 = bf2f(ff[r * (size_t)(2 * HID_) + HID_ + n]);
    float si = x2 / (1.0f + __expf(-x2));
    out[i] = f2bf(x1 * si);
  }
}

// ---------------------------------------------------------------------------
// Generic bf16 GEMM: C[M,N] = A[M,K] @ B[K,N]  (+bias)(+residual), f32/bf16 out
// 256 threads = 8 waves (2m x 4n), block tile 128x128, BK=32.
// Each wave computes 64x32 via eight v_wmma_f32_16x16x32_bf16 per K-chunk.
// A tile staged via global_load_async_to_lds_b128 (ASYNCcnt); B tile staged
// transposed ([n][k]) so all WMMA fragment reads are contiguous ds_load_b128.
// ---------------------------------------------------------------------------
__global__ __launch_bounds__(256) void gemm_bf16_kernel(
    const unsigned short* __restrict__ A, const unsigned short* __restrict__ Bw,
    int M, int N, int K,
    float* __restrict__ outF, unsigned short* __restrict__ outB,
    const float* __restrict__ residual, const float* __restrict__ bias) {
  constexpr int BM = 128, BN = 128, BK = 32, LDA = BK + 8, LDB = BK + 8;
  __shared__ unsigned short As[BM * LDA];
  __shared__ unsigned short Bs[BN * LDB];
  int t = threadIdx.x;
  int L = t & 31, wid = t >> 5;
  int wm = wid & 1, wn = wid >> 1;                 // 2 x 4 wave grid
  int m0 = blockIdx.y * BM, n0 = blockIdx.x * BN;
  int hlf = L >> 4, lidx = L & 15, rlb = hlf * 8;

  V8F acc[4][2];
#pragma unroll
  for (int i = 0; i < 4; i++)
#pragma unroll
    for (int j = 0; j < 2; j++)
#pragma unroll
      for (int v = 0; v < 8; v++) acc[i][j].f[v] = 0.f;

  for (int k0 = 0; k0 < K; k0 += BK) {
    __syncthreads();
    // A tile: 128 x 32, async copy straight into LDS (2 x 16B per thread)
#pragma unroll
    for (int i = 0; i < 2; i++) {
      int u = t + 256 * i;
      int r = u >> 2, g = u & 3;
      async_load_b128(lds_off(&As[r * LDA + g * 8]),
                      A + (size_t)(m0 + r) * K + k0 + g * 8);
    }
    // B tile: 32 x 128, loaded to VGPRs and stored transposed [n][k]
#pragma unroll
    for (int i = 0; i < 2; i++) {
      int u = t + 256 * i;
      int kk = u >> 4, ng = u & 15;
      uint4 val = *(const uint4*)(Bw + (size_t)(k0 + kk) * N + n0 + ng * 8);
      unsigned short tmp[8]; *(uint4*)tmp = val;
#pragma unroll
      for (int j = 0; j < 8; j++) Bs[(ng * 8 + j) * LDB + kk] = tmp[j];
    }
    if (k0 + BK < K) {   // prefetch next tiles -> global_prefetch_b8
      __builtin_prefetch(A + (size_t)(m0 + (t >> 1)) * K + k0 + BK + (t & 1) * 16, 0, 1);
      __builtin_prefetch(Bw + (size_t)(k0 + BK + (t >> 4)) * N + n0 + (t & 15) * 8, 0, 1);
    }
    wait_asynccnt0();        // our async A chunks have landed in LDS
    __syncthreads();         // everyone's chunks have landed

    v16bf af[4], bfr[2];
#pragma unroll
    for (int i = 0; i < 4; i++) {
      int mr = wm * 64 + i * 16 + lidx;
      const v8bf* p0 = (const v8bf*)&As[mr * LDA + hlf * 8];
      const v8bf* p1 = (const v8bf*)&As[mr * LDA + hlf * 8 + 16];
      af[i] = cat16(*p0, *p1);
    }
#pragma unroll
    for (int j = 0; j < 2; j++) {
      int nc = wn * 32 + j * 16 + lidx;
      const v8bf* p0 = (const v8bf*)&Bs[nc * LDB + hlf * 8];
      const v8bf* p1 = (const v8bf*)&Bs[nc * LDB + hlf * 8 + 16];
      bfr[j] = cat16(*p0, *p1);
    }
#pragma unroll
    for (int i = 0; i < 4; i++)
#pragma unroll
      for (int j = 0; j < 2; j++)
        acc[i][j].v = __builtin_amdgcn_wmma_f32_16x16x32_bf16(
            false, af[i], false, bfr[j], (short)0, acc[i][j].v, false, false);
  }

#pragma unroll
  for (int i = 0; i < 4; i++)
#pragma unroll
    for (int j = 0; j < 2; j++) {
      int cbase = n0 + wn * 32 + j * 16 + lidx;
      int rbase = m0 + wm * 64 + i * 16 + rlb;
#pragma unroll
      for (int v = 0; v < 8; v++) {
        int r = rbase + v, c = cbase;
        float val = acc[i][j].f[v];
        if (bias)     val += bias[c];
        if (residual) val += residual[(size_t)r * N + c];
        if (outF) outF[(size_t)r * N + c] = val;
        else      outB[(size_t)r * N + c] = f2bf(val);
      }
    }
}

// ---------------------------------------------------------------------------
// Flash attention (causal, GQA). 128 threads = 4 waves, 16 query rows/wave,
// 32-key tiles. QK^T and PV both via v_wmma_f32_16x16x32_bf16.
// K tile staged via async-to-LDS; V tile staged transposed.
// ---------------------------------------------------------------------------
__global__ __launch_bounds__(128) void attn_kernel(
    const unsigned short* __restrict__ Qb, const unsigned short* __restrict__ Kb,
    const unsigned short* __restrict__ Vb, const int* __restrict__ amask,
    unsigned short* __restrict__ Ob, int S_, int H_, int G_) {
  constexpr int LDK = 72, LDV = 40, LDP = 40;
  __shared__ unsigned short Kt[32 * LDK];      // [key][d]
  __shared__ unsigned short Vt[64 * LDV];      // [d][key] (transposed)
  __shared__ unsigned short Pt[4 * 16 * LDP];  // per-wave P relayout patch

  int t = threadIdx.x, L = t & 31, w = t >> 5;
  int hlf = L >> 4, lidx = L & 15, rlb = hlf * 8;
  int qblk = blockIdx.x;
  int bh = blockIdx.y, b = bh / H_, h = bh % H_;
  int g = h / (H_ / G_);
  int qbase = qblk * 64 + w * 16;
  unsigned short* Pw = Pt + w * 16 * LDP;

  // Load this wave's Q A-fragments (kept in VGPRs for the whole loop)
  v16bf aq[2];
  {
    int qrow = qbase + lidx;
    const unsigned short* qp = Qb + (((size_t)(b * S_ + qrow)) * H_ + h) * 64;
#pragma unroll
    for (int c = 0; c < 2; c++) {
      const v8bf* p0 = (const v8bf*)(qp + c * 32 + rlb);
      const v8bf* p1 = (const v8bf*)(qp + c * 32 + rlb + 16);
      aq[c] = cat16(*p0, *p1);
    }
  }

  float mrow[8], lrow[8];
  V8F acc[4];
#pragma unroll
  for (int v = 0; v < 8; v++) { mrow[v] = -1e30f; lrow[v] = 0.f; }
#pragma unroll
  for (int dt = 0; dt < 4; dt++)
#pragma unroll
    for (int v = 0; v < 8; v++) acc[dt].f[v] = 0.f;

  int nkt = qblk * 2 + 2;   // causal bound for this 64-row block
  for (int kt = 0; kt < nkt; kt++) {
    __syncthreads();
    // K tile: 32 keys x 64 d, async straight into LDS
#pragma unroll
    for (int i = 0; i < 2; i++) {
      int u = t + 128 * i;
      int key = u >> 3, d0 = (u & 7) * 8;
      int kidx = kt * 32 + key;
      async_load_b128(lds_off(&Kt[key * LDK + d0]),
                      Kb + (((size_t)(b * S_ + kidx)) * G_ + g) * 64 + d0);
    }
    // V tile: loaded to VGPRs, stored transposed [d][key]
#pragma unroll
    for (int i = 0; i < 2; i++) {
      int u = t + 128 * i;
      int key = u >> 3, d0 = (u & 7) * 8;
      int kidx = kt * 32 + key;
      uint4 vval = *(const uint4*)(Vb + (((size_t)(b * S_ + kidx)) * G_ + g) * 64 + d0);
      unsigned short tmp[8]; *(uint4*)tmp = vval;
#pragma unroll
      for (int j = 0; j < 8; j++) Vt[(d0 + j) * LDV + key] = tmp[j];
    }
    wait_asynccnt0();
    __syncthreads();

    // scores: 16q x 32k = two 16x16 C tiles, contraction over HD=64
    V8F sfr[2];
#pragma unroll
    for (int j = 0; j < 2; j++) {
#pragma unroll
      for (int v = 0; v < 8; v++) sfr[j].f[v] = 0.f;
      int nc = j * 16 + lidx;
#pragma unroll
      for (int c = 0; c < 2; c++) {
        const v8bf* p0 = (const v8bf*)&Kt[nc * LDK + c * 32 + hlf * 8];
        const v8bf* p1 = (const v8bf*)&Kt[nc * LDK + c * 32 + hlf * 8 + 16];
        v16bf bk = cat16(*p0, *p1);
        sfr[j].v = __builtin_amdgcn_wmma_f32_16x16x32_bf16(
            false, aq[c], false, bk, (short)0, sfr[j].v, false, false);
      }
    }

    int am0 = amask[b * S_ + kt * 32 + lidx];
    int am1 = amask[b * S_ + kt * 32 + 16 + lidx];

    float pj[2][8], scale[8];
#pragma unroll
    for (int v = 0; v < 8; v++) {
      int r = qbase + rlb + v;
      float s0 = sfr[0].f[v] * 0.125f, s1 = sfr[1].f[v] * 0.125f;
      int k0i = kt * 32 + lidx, k1i = k0i + 16;
      if (k0i > r || !am0) s0 = -1e30f;
      if (k1i > r || !am1) s1 = -1e30f;
      float tmax = fmaxf(s0, s1);
#pragma unroll
      for (int off = 8; off >= 1; off >>= 1)
        tmax = fmaxf(tmax, __shfl_xor(tmax, off, 32));
      float mn = fmaxf(mrow[v], tmax);
      float p0 = __expf(s0 - mn), p1 = __expf(s1 - mn);
      float rs = p0 + p1;
#pragma unroll
      for (int off = 8; off >= 1; off >>= 1)
        rs += __shfl_xor(rs, off, 32);
      scale[v] = __expf(mrow[v] - mn);
      lrow[v] = lrow[v] * scale[v] + rs;
      mrow[v] = mn;
      pj[0][v] = p0; pj[1][v] = p1;
    }

#pragma unroll
    for (int dt = 0; dt < 4; dt++)
#pragma unroll
      for (int v = 0; v < 8; v++) acc[dt].f[v] *= scale[v];

    // Re-layout P (C-layout -> A-layout) through per-wave LDS patch
#pragma unroll
    for (int j = 0; j < 2; j++)
#pragma unroll
      for (int v = 0; v < 8; v++)
        Pw[(rlb + v) * LDP + j * 16 + lidx] = f2bf(pj[j][v]);
    v16bf ap;
    {
      const v8bf* p0 = (const v8bf*)&Pw[lidx * LDP + rlb];
      const v8bf* p1 = (const v8bf*)&Pw[lidx * LDP + rlb + 16];
      ap = cat16(*p0, *p1);
    }

    // O += P @ V  (four 16x16 d-tiles)
#pragma unroll
    for (int dt = 0; dt < 4; dt++) {
      const v8bf* p0 = (const v8bf*)&Vt[(dt * 16 + lidx) * LDV + hlf * 8];
      const v8bf* p1 = (const v8bf*)&Vt[(dt * 16 + lidx) * LDV + hlf * 8 + 16];
      v16bf bv = cat16(*p0, *p1);
      acc[dt].v = __builtin_amdgcn_wmma_f32_16x16x32_bf16(
          false, ap, false, bv, (short)0, acc[dt].v, false, false);
    }
  }

  // write attn output [b][q][h][d] as bf16 (input to O-proj GEMM)
#pragma unroll
  for (int dt = 0; dt < 4; dt++)
#pragma unroll
    for (int v = 0; v < 8; v++) {
      int r = qbase + rlb + v;
      int d = dt * 16 + lidx;
      float val = acc[dt].f[v] / lrow[v];
      Ob[(((size_t)(b * S_ + r)) * H_ + h) * 64 + d] = f2bf(val);
    }
}

// ---------------------------------------------------------------------------
// Host-side orchestration
// ---------------------------------------------------------------------------
extern "C" void kernel_launch(void* const* d_in, const int* in_sizes, int n_in,
                              void* d_out, int out_size, void* d_ws, size_t ws_size,
                              hipStream_t stream) {
  (void)in_sizes; (void)n_in; (void)out_size; (void)ws_size;
  const int Bc = 2, Sc = 2048, Dc = 1024, Hc = 16, Gc = 8, HIDc = 4096;
  const int Mr = Bc * Sc;                     // 4096 tokens

  const float* M     = (const float*)d_in[0];
  const int*   amask = (const int*)d_in[1];
  const float* W_Q   = (const float*)d_in[2];
  const float* W_K   = (const float*)d_in[3];
  const float* W_V   = (const float*)d_in[4];
  const float* W_O   = (const float*)d_in[5];
  const float* W1_w  = (const float*)d_in[6];
  const float* W1_b  = (const float*)d_in[7];
  const float* W2_w  = (const float*)d_in[8];
  const float* W2_b  = (const float*)d_in[9];
  float* out = (float*)d_out;

  char* ws = (char*)d_ws;
  size_t off = 0;
  auto alloc = [&](size_t bytes) { size_t o = off; off += (bytes + 255) & ~(size_t)255; return o; };

  unsigned short* bwq  = (unsigned short*)(ws + alloc((size_t)Dc * Dc * 2));
  unsigned short* bwk  = (unsigned short*)(ws + alloc((size_t)Dc * Gc * 64 * 2));
  unsigned short* bwv  = (unsigned short*)(ws + alloc((size_t)Dc * Gc * 64 * 2));
  unsigned short* bwo  = (unsigned short*)(ws + alloc((size_t)Dc * Dc * 2));
  unsigned short* bw1  = (unsigned short*)(ws + alloc((size_t)Dc * 2 * HIDc * 2));
  unsigned short* bw2  = (unsigned short*)(ws + alloc((size_t)HIDc * Dc * 2));
  unsigned short* mnb  = (unsigned short*)(ws + alloc((size_t)Mr * Dc * 2));
  float*          qf   = (float*)(ws + alloc((size_t)Mr * Dc * 4));
  float*          kf   = (float*)(ws + alloc((size_t)Mr * Gc * 64 * 4));
  float*          vf   = (float*)(ws + alloc((size_t)Mr * Gc * 64 * 4));
  unsigned short* qb   = (unsigned short*)(ws + alloc((size_t)Mr * Dc * 2));
  unsigned short* kb   = (unsigned short*)(ws + alloc((size_t)Mr * Gc * 64 * 2));
  unsigned short* vb   = (unsigned short*)(ws + alloc((size_t)Mr * Gc * 64 * 2));
  unsigned short* attb = (unsigned short*)(ws + alloc((size_t)Mr * Dc * 2));
  float*          m2   = (float*)(ws + alloc((size_t)Mr * Dc * 4));
  unsigned short* ffb  = (unsigned short*)(ws + alloc((size_t)Mr * 2 * HIDc * 2));
  unsigned short* swib = (unsigned short*)(ws + alloc((size_t)Mr * HIDc * 2));

  // 1) Weights -> bf16
  auto cvt = [&](const float* s, unsigned short* d, size_t n) {
    f32_to_bf16_kernel<<<2048, 256, 0, stream>>>(s, d, n);
  };
  cvt(W_Q, bwq, (size_t)Dc * Dc);
  cvt(W_K, bwk, (size_t)Dc * Gc * 64);
  cvt(W_V, bwv, (size_t)Dc * Gc * 64);
  cvt(W_O, bwo, (size_t)Dc * Dc);
  cvt(W1_w, bw1, (size_t)Dc * 2 * HIDc);
  cvt(W2_w, bw2, (size_t)HIDc * Dc);

  // 2) RMSNorm(M) -> bf16
  rmsnorm_bf16_kernel<<<Mr, 256, 0, stream>>>(M, mnb, Dc);

  // 3) Q/K/V projections (bf16 WMMA GEMMs, f32 out)
  gemm_bf16_kernel<<<dim3(Dc / 128, Mr / 128), 256, 0, stream>>>(
      mnb, bwq, Mr, Dc, Dc, qf, nullptr, nullptr, nullptr);
  gemm_bf16_kernel<<<dim3((Gc * 64) / 128, Mr / 128), 256, 0, stream>>>(
      mnb, bwk, Mr, Gc * 64, Dc, kf, nullptr, nullptr, nullptr);
  gemm_bf16_kernel<<<dim3((Gc * 64) / 128, Mr / 128), 256, 0, stream>>>(
      mnb, bwv, Mr, Gc * 64, Dc, vf, nullptr, nullptr, nullptr);

  // 4) RoPE(Q), RoPE(K), convert V -> bf16
  rope_bf16_kernel<<<(Mr * Hc) / 8, 256, 0, stream>>>(qf, qb, Hc, Sc);
  rope_bf16_kernel<<<(Mr * Gc) / 8, 256, 0, stream>>>(kf, kb, Gc, Sc);
  cvt(vf, vb, (size_t)Mr * Gc * 64);

  // 5) Causal GQA flash attention (WMMA)
  attn_kernel<<<dim3(Sc / 64, Bc * Hc), 128, 0, stream>>>(
      qb, kb, vb, amask, attb, Sc, Hc, Gc);

  // 6) O-projection + residual -> m2 (f32)
  gemm_bf16_kernel<<<dim3(Dc / 128, Mr / 128), 256, 0, stream>>>(
      attb, bwo, Mr, Dc, Dc, m2, nullptr, M, nullptr);

  // 7) RMSNorm(m2) -> bf16 (reuse mnb)
  rmsnorm_bf16_kernel<<<Mr, 256, 0, stream>>>(m2, mnb, Dc);

  // 8) FFN up (bias fused, bf16 out)
  gemm_bf16_kernel<<<dim3((2 * HIDc) / 128, Mr / 128), 256, 0, stream>>>(
      mnb, bw1, Mr, 2 * HIDc, Dc, nullptr, ffb, nullptr, W1_b);

  // 9) SwiGLU
  swiglu_kernel<<<8192, 256, 0, stream>>>(ffb, swib, HIDc, (size_t)Mr * HIDc);

  // 10) FFN down + bias + residual -> final f32 output
  gemm_bf16_kernel<<<dim3(Dc / 128, Mr / 128), 256, 0, stream>>>(
      swib, bw2, Mr, Dc, HIDc, out, nullptr, m2, W2_b);
}
